// Model_18468359373243
// MI455X (gfx1250) — matrix operand order repaired and checked by reading, not compiled, not run on hardware
//
#include <hip/hip_runtime.h>
#include <cstdint>
#include <cstddef>

#define B_    32
#define T_    1024
#define IN_   32
#define H_    512
#define G4_   2048
#define OUT_  32
#define UNITS 16     // hidden units per workgroup
#define SLICE 64     // gate rows per workgroup (4 gates * UNITS)
#define NWGL  32     // workgroups per layer
#define NWGT  64     // total persistent workgroups (2 layers pipelined)

typedef __attribute__((ext_vector_type(8)))  __bf16 v8bf;
typedef __attribute__((ext_vector_type(16))) __bf16 v16bf;
typedef __attribute__((ext_vector_type(8)))  float  v8f;

// ---------------- CDNA5 async global->LDS staging (ASYNCcnt path) ----------------
#if defined(__has_builtin)
#  if __has_builtin(__builtin_amdgcn_global_load_async_to_lds_b128)
#    define HAVE_ASYNC_LDS 1
#  endif
#endif

typedef int v4i_gcc __attribute__((vector_size(16)));
typedef __attribute__((address_space(1))) v4i_gcc g_v4i;   // global
typedef __attribute__((address_space(3))) v4i_gcc l_v4i;   // LDS

static __device__ __forceinline__ void copy16_g2l(__bf16* ldst, const __bf16* gsrc) {
#ifdef HAVE_ASYNC_LDS
  __builtin_amdgcn_global_load_async_to_lds_b128((g_v4i*)(void*)gsrc,
                                                 (l_v4i*)(void*)ldst, 0, 0);
#else
  *(v8bf*)ldst = *(const v8bf*)gsrc;
#endif
}

static __device__ __forceinline__ void async_fence() {
#ifdef HAVE_ASYNC_LDS
#  if __has_builtin(__builtin_amdgcn_s_wait_asynccnt)
  __builtin_amdgcn_s_wait_asynccnt(0);
#  else
  asm volatile("s_wait_asynccnt 0" ::: "memory");
#  endif
#endif
}

// ---------------- WMMA fragment loaders (CDNA5 ISA §7.12.2 layouts) ----------------
// A matrix 16x32 bf16, row-major source: lane r=lane&15 holds row (row0+r);
// lanes 0-15 carry K=[k0..k0+8) and [k0+16..k0+24), lanes 16-31 the other 8-chunks.
static __device__ __forceinline__ v16bf load_a_frag(const __bf16* base, int row0, int ld,
                                                    int k0, int lane) {
  const int r = lane & 15, hs = lane >> 4;
  const __bf16* p = base + (size_t)(row0 + r) * (size_t)ld + k0 + hs * 8;
  v8bf lo = *(const v8bf*)p;
  v8bf hi = *(const v8bf*)(p + 16);
  v16bf f;
#pragma unroll
  for (int i = 0; i < 8; ++i) { f[i] = lo[i]; f[i + 8] = hi[i]; }
  return f;
}

// B matrix 32x16 bf16: column n = lane&15 (== weight row n0+n for B = W^T);
// lanes 0-15 carry K=[k0..k0+8) and [k0+16..k0+24), lanes 16-31 the other 8-chunks.
static __device__ __forceinline__ v16bf load_b_frag(const __bf16* base, int n0, int ld,
                                                    int k0, int lane) {
  const int r = lane & 15, hs = lane >> 4;
  const __bf16* p = base + (size_t)(n0 + r) * (size_t)ld + k0 + hs * 8;
  v8bf lo = *(const v8bf*)p;
  v8bf hi = *(const v8bf*)(p + 16);
  v16bf f;
#pragma unroll
  for (int i = 0; i < 8; ++i) { f[i] = lo[i]; f[i + 8] = hi[i]; }
  return f;
}

static __device__ __forceinline__ v8f wmma_bf16(v16bf a, v16bf b, v8f c) {
  return __builtin_amdgcn_wmma_f32_16x16x32_bf16(false, a, false, b, (short)0, c, false, false);
}

static __device__ __forceinline__ float fsigmoid(float x) {
  return 1.0f / (1.0f + __expf(-x));
}
static __device__ __forceinline__ float ftanh_fast(float x) {
  return 1.0f - 2.0f / (1.0f + __expf(2.0f * x));
}

// Grid-wide barrier for the persistent pipeline (64 co-resident WGs).
static __device__ __forceinline__ void grid_barrier(unsigned* cnt, unsigned target) {
  __syncthreads();
  if (threadIdx.x == 0) {
    __threadfence();
    __atomic_fetch_add(cnt, 1u, __ATOMIC_RELAXED);
    while (__atomic_load_n(cnt, __ATOMIC_RELAXED) < target) {
      __builtin_amdgcn_s_sleep(1);
    }
    __threadfence();
  }
  __syncthreads();
}

// ---------------- small utility kernels ----------------
__global__ void zero_kernel(unsigned* p, int n) {
  int i = blockIdx.x * blockDim.x + threadIdx.x;
  if (i < n) p[i] = 0u;
}

__global__ void cvt_kernel(const float* __restrict__ src, __bf16* __restrict__ dst, int n) {
  int i = blockIdx.x * blockDim.x + threadIdx.x;
  if (i < n) dst[i] = (__bf16)src[i];
}

// ---------------- fused 2-layer pipelined LSTM ----------------
// 64 persistent WGs. WGs [0,32): layer 0 at step s; WGs [32,64): layer 1 at step s-1.
// Each WG owns 16 hidden units: its 64 gate rows of W_hh live in LDS (bf16, 64 KB).
// h_{t-1} is staged into LDS (async) each round; h ping-pongs through global bf16.
// Layer 1 additionally does its input GEMM (h1[t] @ W_ih1^T) on the fly from L2.
__global__ __launch_bounds__(256, 1)
void lstm_pipe_kernel(const __bf16* __restrict__ xbf,      // [B,T,IN] bf16
                      const float*  __restrict__ Wih0_f32, // [4H,IN]
                      const float*  __restrict__ Whh0_f32, // [4H,H]
                      const float*  __restrict__ bih0,
                      const float*  __restrict__ bhh0,
                      const __bf16* __restrict__ Wih1bf,   // [4H,H] bf16
                      const float*  __restrict__ Whh1_f32, // [4H,H]
                      const float*  __restrict__ bih1,
                      const float*  __restrict__ bhh1,
                      __bf16* __restrict__ hbuf0,          // [2][B][H]
                      __bf16* __restrict__ hbuf1,          // [2][B][H]
                      __bf16* __restrict__ h1seq,          // [B,T,H]
                      __bf16* __restrict__ h2seq,          // [B,T,H]
                      unsigned* __restrict__ cnt) {
  __shared__ __bf16 sW[SLICE][H_];     // 64 KB recurrent weights (region-gathered)
  __shared__ __bf16 sWi[SLICE][IN_];   // 4 KB  (layer-0 input weights)
  __shared__ __bf16 sH[B_][H_];        // 32 KB staged h_{t-1}
  __shared__ float  sGates[B_][SLICE]; // 8 KB
  __shared__ float  sBias[SLICE];

  const int tid   = threadIdx.x;
  const int role  = blockIdx.x >> 5;      // 0 = layer0, 1 = layer1
  const int wg    = blockIdx.x & 31;
  const int unit0 = wg * UNITS;

  const float* Whh = role ? Whh1_f32 : Whh0_f32;
  const float* bi  = role ? bih1 : bih0;
  const float* bh  = role ? bhh1 : bhh0;
  __bf16* hbuf     = role ? hbuf1 : hbuf0;
  __bf16* hseqOut  = role ? h2seq : h1seq;

  // Stage W_hh slice (f32 -> bf16) into LDS; rows gathered as i/f/g/o blocks of 16.
  for (int idx = tid; idx < SLICE * H_; idx += 256) {
    int lr = idx >> 9;          // / H_
    int k  = idx & (H_ - 1);
    int grow = (lr >> 4) * H_ + unit0 + (lr & 15);
    sW[lr][k] = (__bf16)Whh[(size_t)grow * H_ + k];
  }
  if (role == 0) {
    for (int idx = tid; idx < SLICE * IN_; idx += 256) {
      int lr = idx >> 5;
      int k  = idx & (IN_ - 1);
      int grow = (lr >> 4) * H_ + unit0 + (lr & 15);
      sWi[lr][k] = (__bf16)Wih0_f32[(size_t)grow * IN_ + k];
    }
  }
  if (tid < SLICE) {
    int grow = (tid >> 4) * H_ + unit0 + (tid & 15);
    sBias[tid] = bi[grow] + bh[grow];
  }
  // zero our unit slice of both h ping-pong buffers (h0 = 0)
  for (int p = tid; p < 2 * B_ * UNITS; p += 256) {
    int buf = p >> 9;
    int m = (p >> 4) & 31;
    int j = p & 15;
    hbuf[(size_t)buf * B_ * H_ + (size_t)m * H_ + unit0 + j] = (__bf16)0.0f;
  }
  float creg[2] = {0.f, 0.f};  // cell state: 2 (m,j) pairs per thread

  grid_barrier(cnt, NWGT);

  const int wave  = tid >> 5;
  const int lane  = tid & 31;
  const int ntile = wave & 3;          // 4 N-tiles of 16 gate rows
  const int m0    = (wave >> 2) * 16;  // 2 M-tiles of 16 batch rows
  const int n0    = ntile * 16;

  for (int s = 0; s <= T_; ++s) {
    const bool active = (role == 0) ? (s < T_) : (s >= 1);
    if (active) {
      const int t = (role == 0) ? s : (s - 1);
      const __bf16* cur = hbuf + (size_t)(t & 1) * B_ * H_;
      __bf16* nxt = hbuf + (size_t)((t + 1) & 1) * B_ * H_;

      // stage h_{t-1} [32x512 bf16 = 32 KB] into LDS via async copies
      for (int ch = tid; ch < (B_ * H_) / 8; ch += 256) {
        copy16_g2l(&sH[0][0] + ch * 8, cur + ch * 8);
      }
      async_fence();
      __syncthreads();

      v8f acc = {};
      if (role == 0) {
        // x_t @ W_ih0^T (K = 32, single WMMA)
        v16bf bx = load_b_frag(&sWi[0][0], n0, IN_, 0, lane);
        v16bf ax = load_a_frag(xbf + (size_t)t * IN_, m0, T_ * IN_, 0, lane);
        acc = wmma_bf16(ax, bx, acc);
      } else {
        // h1[t] @ W_ih1^T on the fly (weights from L2-resident bf16 copy)
        const __bf16* Ain = h1seq + (size_t)t * H_;
        const int gn0 = ntile * H_ + unit0;  // raw gate-row base in W_ih1
#pragma unroll 4
        for (int k0 = 0; k0 < H_; k0 += 32) {
          v16bf b = load_b_frag(Wih1bf, gn0, H_, k0, lane);
          v16bf a = load_a_frag(Ain, m0, T_ * H_, k0, lane);
          acc = wmma_bf16(a, b, acc);
        }
      }
      // recurrent h_{t-1} @ W_hh^T, all operands from LDS
#pragma unroll 4
      for (int k0 = 0; k0 < H_; k0 += 32) {
        v16bf b = load_b_frag(&sW[0][0], n0, H_, k0, lane);
        v16bf a = load_a_frag(&sH[0][0], m0, H_, k0, lane);
        acc = wmma_bf16(a, b, acc);
      }
      // scatter accumulator -> LDS gate buffer [batch][64]
      {
        int col = n0 + (lane & 15);
        int mr  = m0 + (lane >> 4) * 8;
#pragma unroll
        for (int i = 0; i < 8; ++i) sGates[mr + i][col] = acc[i];
      }
      __syncthreads();
      // elementwise LSTM cell: each thread owns 2 (m, j) pairs
#pragma unroll
      for (int q = 0; q < 2; ++q) {
        int p = tid + q * 256;
        int m = p >> 4;
        int j = p & 15;
        float gi = sGates[m][j]      + sBias[j];
        float gf = sGates[m][16 + j] + sBias[16 + j];
        float gg = sGates[m][32 + j] + sBias[32 + j];
        float go = sGates[m][48 + j] + sBias[48 + j];
        float iv = fsigmoid(gi);
        float fv = fsigmoid(gf);
        float gv = ftanh_fast(gg);
        float ov = fsigmoid(go);
        float c = fv * creg[q] + iv * gv;
        creg[q] = c;
        float h = ov * ftanh_fast(c);
        __bf16 hb = (__bf16)h;
        nxt[(size_t)m * H_ + unit0 + j] = hb;
        hseqOut[((size_t)m * T_ + t) * H_ + unit0 + j] = hb;
      }
    }
    grid_barrier(cnt, (unsigned)(s + 2) * NWGT);
  }
}

// ---------------- decoder: out = h2 @ W_dec^T + b_dec ----------------
__global__ __launch_bounds__(256)
void dec_kernel(const __bf16* __restrict__ h2seq,
                const __bf16* __restrict__ Wdecbf,
                const float* __restrict__ bdec,
                float* __restrict__ out) {
  const int t = blockIdx.x;
  const int wave = threadIdx.x >> 5;
  if (wave >= 4) return;  // 2 M-tiles x 2 N-tiles = 4 waves do the work
  const int lane = threadIdx.x & 31;
  const int m0 = (wave >> 1) * 16;
  const int n0 = (wave & 1) * 16;
  const __bf16* A = h2seq + (size_t)t * H_;
  v8f acc = {};
#pragma unroll 4
  for (int k0 = 0; k0 < H_; k0 += 32) {
    v16bf b = load_b_frag(Wdecbf, n0, H_, k0, lane);
    v16bf a = load_a_frag(A, m0, T_ * H_, k0, lane);
    acc = wmma_bf16(a, b, acc);
  }
  const int col = n0 + (lane & 15);
  const int mr  = m0 + (lane >> 4) * 8;
  const float bb = bdec[col];
#pragma unroll
  for (int i = 0; i < 8; ++i) {
    out[((size_t)(mr + i) * T_ + t) * OUT_ + col] = acc[i] + bb;
  }
}

// ---------------- host launch ----------------
extern "C" void kernel_launch(void* const* d_in, const int* in_sizes, int n_in,
                              void* d_out, int out_size, void* d_ws, size_t ws_size,
                              hipStream_t stream) {
  (void)in_sizes; (void)n_in; (void)out_size; (void)ws_size;
  const float* x     = (const float*)d_in[0];
  const float* W_ih0 = (const float*)d_in[1];
  const float* W_hh0 = (const float*)d_in[2];
  const float* b_ih0 = (const float*)d_in[3];
  const float* b_hh0 = (const float*)d_in[4];
  const float* W_ih1 = (const float*)d_in[5];
  const float* W_hh1 = (const float*)d_in[6];
  const float* b_ih1 = (const float*)d_in[7];
  const float* b_hh1 = (const float*)d_in[8];
  const float* W_dec = (const float*)d_in[9];
  const float* b_dec = (const float*)d_in[10];

  char* w = (char*)d_ws;
  auto take = [&](size_t bytes) -> char* {
    char* p = w;
    w += (bytes + 255) & ~(size_t)255;
    return p;
  };
  unsigned* counters = (unsigned*)take(2 * sizeof(unsigned));
  __bf16* xbf    = (__bf16*)take((size_t)B_ * T_ * IN_ * 2);   // 2 MB
  __bf16* Wih1bf = (__bf16*)take((size_t)G4_ * H_ * 2);        // 2 MB
  __bf16* Wdecbf = (__bf16*)take((size_t)OUT_ * H_ * 2);       // 32 KB
  __bf16* hbuf0  = (__bf16*)take((size_t)2 * B_ * H_ * 2);     // 64 KB
  __bf16* hbuf1  = (__bf16*)take((size_t)2 * B_ * H_ * 2);     // 64 KB
  __bf16* h1seq  = (__bf16*)take((size_t)B_ * T_ * H_ * 2);    // 32 MB
  __bf16* h2seq  = (__bf16*)take((size_t)B_ * T_ * H_ * 2);    // 32 MB

  zero_kernel<<<1, 64, 0, stream>>>(counters, 2);
  cvt_kernel<<<(B_ * T_ * IN_ + 255) / 256, 256, 0, stream>>>(x, xbf, B_ * T_ * IN_);
  cvt_kernel<<<(G4_ * H_ + 255) / 256, 256, 0, stream>>>(W_ih1, Wih1bf, G4_ * H_);
  cvt_kernel<<<(OUT_ * H_ + 255) / 256, 256, 0, stream>>>(W_dec, Wdecbf, OUT_ * H_);

  lstm_pipe_kernel<<<NWGT, 256, 0, stream>>>(xbf, W_ih0, W_hh0, b_ih0, b_hh0,
                                             Wih1bf, W_hh1, b_ih1, b_hh1,
                                             hbuf0, hbuf1, h1seq, h2seq, counters);

  dec_kernel<<<T_, 256, 0, stream>>>(h2seq, Wdecbf, b_dec, (float*)d_out);
}
